// RecurrentActorCritic_88923002896421
// MI455X (gfx1250) — compile-verified, matrix operand order checked by reading, not run on hardware
//
#include <hip/hip_runtime.h>
#include <hip/hip_bf16.h>
#include <stdint.h>

// Fused recurrent actor-critic for MI455X (gfx1250, wave32, WMMA bf16).
// Grid: B/16 blocks x 256 threads (8 waves). Decoupled pipeline per block:
//   waves 0-3 (GRU group): gate GEMMs + h update only (the true serial chain)
//   waves 4-7 (head group): async-DMA x tiles, produce feat(t) up to 4 steps
//     AHEAD (ring), consume h(t-1) one step BEHIND for heads/logits/values.
// Cross-group sync: monotonic LDS counters (ds_add release / ds_load acquire
// polling with s_sleep) -- no workgroup barriers on the recurrence.

#define T_DIM 512
#define B_DIM 2048
#define OBS_D 64
#define GH_D  64
#define A_D   16

typedef __attribute__((ext_vector_type(16))) __bf16       v16bf;
typedef __attribute__((ext_vector_type(8)))  float        v8f;
typedef __attribute__((ext_vector_type(8)))  unsigned int v8u;

// K index pattern for 16-bit A/B fragments (CDNA5 ISA, 16-bit A 16x32):
// lanes 0-15: VGPR0..3 -> K=0..7, VGPR4..7 -> K=16..23 ; lanes 16-31: +8
__device__ __forceinline__ int kpat(int v, int hi) {
  return ((v & 4) ? 16 : 0) + 2 * (v & 3) + (hi ? 8 : 0);
}

__device__ __forceinline__ v16bf frag_lds(const __bf16* s, int row, int hi,
                                          int kbase, int stride) {
  v16bf f;
#pragma unroll
  for (int v = 0; v < 8; ++v) {
    int k = kbase + kpat(v, hi);
    f[2 * v]     = s[row * stride + k];
    f[2 * v + 1] = s[row * stride + k + 1];
  }
  return f;
}

__device__ __forceinline__ v16bf frag_lds_f32(const float* s, int row, int hi,
                                              int kbase, int stride) {
  v16bf f;
#pragma unroll
  for (int v = 0; v < 8; ++v) {
    int k = kbase + kpat(v, hi);
    f[2 * v]     = (__bf16)s[row * stride + k];
    f[2 * v + 1] = (__bf16)s[row * stride + k + 1];
  }
  return f;
}

__device__ __forceinline__ v16bf frag_gf32(const float* W, int row, int hi,
                                           int kbase) {
  v16bf f;
#pragma unroll
  for (int v = 0; v < 8; ++v) {
    int k = kbase + kpat(v, hi);
    f[2 * v]     = (__bf16)W[row * 64 + k];
    f[2 * v + 1] = (__bf16)W[row * 64 + k + 1];
  }
  return f;
}

__device__ __forceinline__ v8f wmma_bf16(v16bf a, v16bf b, v8f c) {
  return __builtin_amdgcn_wmma_f32_16x16x32_bf16(
      false, a, false, b, (short)0, c, false, false);
}

// Native gfx1250 TRANS tanh; v_nop covers the TRANS result-use hazard.
__device__ __forceinline__ float fast_tanh(float x) {
  float r;
  asm volatile("v_tanh_f32 %0, %1\n\tv_nop" : "=v"(r) : "v"(x));
  return r;
}
__device__ __forceinline__ float fast_sigmoid(float x) {
  return __builtin_fmaf(0.5f, fast_tanh(0.5f * x), 0.5f);
}

// CDNA5 async DMA: one head wave moves its 1KB quarter of a 4KB x tile.
__device__ __forceinline__ void issue_x_quarter(const float* xsrc, float* ldsdst,
                                                int hw, int lane) {
#pragma unroll
  for (int j = 0; j < 2; ++j) {
    int off = hw * 256 + (j * 32 + lane) * 4;  // float idx; 16B per lane/issue
    unsigned int ldsa = (unsigned int)(uintptr_t)(ldsdst + off);
    asm volatile("global_load_async_to_lds_b128 %0, %1, off"
                 :: "v"(ldsa), "v"(xsrc + off) : "memory");
  }
}
__device__ __forceinline__ void wait_async0() {
  asm volatile("s_wait_asynccnt 0x0" ::: "memory");
}

// Monotonic LDS counters. LDS ops are in-order per wave (DScnt), so a
// release-add publishes all prior tile stores; acquire-poll sees them.
__device__ __forceinline__ void wait_ge(int* ctr, int target) {
  while (__hip_atomic_load(ctr, __ATOMIC_ACQUIRE,
                           __HIP_MEMORY_SCOPE_WORKGROUP) < target)
    __builtin_amdgcn_s_sleep(1);
}
__device__ __forceinline__ void publish(int* ctr, int lane) {
  if (lane == 0)
    __hip_atomic_fetch_add(ctr, 1, __ATOMIC_RELEASE,
                           __HIP_MEMORY_SCOPE_WORKGROUP);
}

__global__ __launch_bounds__(256, 1)
void rac_fused_kernel(const float* __restrict__ x, const float* __restrict__ hx,
                      const unsigned char* __restrict__ dones,  // numpy bool: 1B
                      const float* __restrict__ base_w, const float* __restrict__ base_b,
                      const float* __restrict__ w_ih, const float* __restrict__ w_hh,
                      const float* __restrict__ b_ih, const float* __restrict__ b_hh,
                      const float* __restrict__ aw1, const float* __restrict__ ab1,
                      const float* __restrict__ aw2, const float* __restrict__ ab2,
                      const float* __restrict__ cw1, const float* __restrict__ cb1,
                      const float* __restrict__ cw2, const float* __restrict__ cb2,
                      float* __restrict__ out_logits, float* __restrict__ out_values,
                      float* __restrict__ out_h) {
  __shared__ float  xsf[2][16 * 64];        // async DMA landing buffers (8KB)
  __shared__ __bf16 featR[4][16 * 64];      // feat ring, depth 4 (8KB)
  __shared__ __bf16 hR[4][16 * 64];         // h ring, depth 4 (8KB); slot t&3
  __shared__ __bf16 as1R[2][16 * 64];       // actor hidden ring (4KB)
  __shared__ __bf16 cs1R[2][16 * 64];       // critic hidden ring (4KB)
  __shared__ unsigned int dbits[T_DIM / 2]; // done flags, 1 bit each (1KB)
  __shared__ int c_x, c_feat, c_h, c_hd, c_out;

  const int tid   = threadIdx.x;
  const int wv    = tid >> 5;
  const int lane  = tid & 31;
  const int hi    = (lane >> 4) & 1;
  const int cl    = lane & 15;
  const int bbase = blockIdx.x * 16;

  // ---- prologue: bit-pack done flags, stage initial h, zero counters ----
  for (int w = tid; w < T_DIM / 2; w += 256) {
    unsigned int word = 0;
#pragma unroll
    for (int half = 0; half < 2; ++half) {
      const unsigned char* p = dones + (size_t)(w * 2 + half) * B_DIM + bbase;
#pragma unroll
      for (int r = 0; r < 16; ++r)
        word |= (p[r] ? 1u : 0u) << (half * 16 + r);
    }
    dbits[w] = word;
  }
  for (int i = tid; i < 16 * 64; i += 256)
    hR[3][i] = (__bf16)hx[(size_t)(bbase + (i >> 6)) * GH_D + (i & 63)];
  if (tid == 0) { c_x = 0; c_feat = 0; c_h = 0; c_hd = 0; c_out = 0; }
  __syncthreads();  // the ONLY block-wide barrier

  if (wv < 4) {
    // ================= GRU group: the serial recurrence =================
    const int gcol = wv * 16 + cl;
    v16bf Bir[2], Biz[2], Bin[2], Bhr[2], Bhz[2], Bhn[2];
#pragma unroll
    for (int kk = 0; kk < 2; ++kk) {
      Bir[kk] = frag_gf32(w_ih, gcol,       hi, kk * 32);
      Biz[kk] = frag_gf32(w_ih, 64 + gcol,  hi, kk * 32);
      Bin[kk] = frag_gf32(w_ih, 128 + gcol, hi, kk * 32);
      Bhr[kk] = frag_gf32(w_hh, gcol,       hi, kk * 32);
      Bhz[kk] = frag_gf32(w_hh, 64 + gcol,  hi, kk * 32);
      Bhn[kk] = frag_gf32(w_hh, 128 + gcol, hi, kk * 32);
    }
    const float bias_r  = b_ih[gcol]      + b_hh[gcol];
    const float bias_z  = b_ih[64 + gcol] + b_hh[64 + gcol];
    const float bias_xn = b_ih[128 + gcol];
    const float bias_hn = b_hh[128 + gcol];

    for (int t = 0; t < T_DIM; ++t) {
      wait_ge(&c_feat, 4 * (t + 1));          // feat(t) produced (head group)
      wait_ge(&c_h, 4 * t);                   // h(t-1) from all GRU waves
      if (t >= 4) wait_ge(&c_hd, 4 * (t - 3)); // h-ring slot t&3 consumed

      const __bf16* fsl = featR[t & 3];
      const __bf16* hsl = hR[(t - 1) & 3];
      const unsigned int word = dbits[t >> 1];
      const int sh = (t & 1) * 16;
      const unsigned int amask = ((word >> (sh + cl)) & 1u) ? 0u : 0xFFFFFFFFu;

      v16bf fA[2], hA[2];
#pragma unroll
      for (int kk = 0; kk < 2; ++kk) {
        fA[kk] = frag_lds(fsl, cl, hi, kk * 32, 64);
        v8u hu = __builtin_bit_cast(v8u, frag_lds(hsl, cl, hi, kk * 32, 64));
        hu = hu & amask;                      // done-reset: zero whole rows
        hA[kk] = __builtin_bit_cast(v16bf, hu);
      }
      float hp[8];
#pragma unroll
      for (int i = 0; i < 8; ++i) {
        int row = i + hi * 8;
        hp[i] = ((word >> (sh + row)) & 1u) ? 0.0f
                                            : (float)hsl[row * 64 + gcol];
      }
      v8f ar, az, axn, ahn;
#pragma unroll
      for (int i = 0; i < 8; ++i) {
        ar[i] = bias_r; az[i] = bias_z; axn[i] = bias_xn; ahn[i] = bias_hn;
      }
#pragma unroll
      for (int kk = 0; kk < 2; ++kk) {
        ar  = wmma_bf16(fA[kk], Bir[kk], ar);
        ar  = wmma_bf16(hA[kk], Bhr[kk], ar);
        az  = wmma_bf16(fA[kk], Biz[kk], az);
        az  = wmma_bf16(hA[kk], Bhz[kk], az);
        axn = wmma_bf16(fA[kk], Bin[kk], axn);
        ahn = wmma_bf16(hA[kk], Bhn[kk], ahn);
      }
      __bf16* hdst = hR[t & 3];
#pragma unroll
      for (int i = 0; i < 8; ++i) {
        int row = i + hi * 8;
        float r = fast_sigmoid(ar[i]);
        float z = fast_sigmoid(az[i]);
        float n = fast_tanh(axn[i] + r * ahn[i]);
        hdst[row * 64 + gcol] = (__bf16)((1.0f - z) * n + z * hp[i]);
      }
      publish(&c_h, lane);                    // release: h stores visible
    }
    // h_last [1,B,GH]: h(511) lives in slot 511&3 == 3
    wait_ge(&c_h, 4 * T_DIM);
    for (int i = tid; i < 16 * 64; i += 128)
      out_h[(size_t)(bbase + (i >> 6)) * GH_D + (i & 63)] = (float)hR[3][i];
  } else {
    // ============ head group: feat runs AHEAD, heads run BEHIND ==========
    const int hw   = wv - 4;
    const int gcol = hw * 16 + cl;
    v16bf Bbase[2], Ba1[2], Bc1[2], Bo[2];
#pragma unroll
    for (int kk = 0; kk < 2; ++kk) {
      Bbase[kk] = frag_gf32(base_w, gcol, hi, kk * 32);
      Ba1[kk]   = frag_gf32(aw1,    gcol, hi, kk * 32);
      Bc1[kk]   = frag_gf32(cw1,    gcol, hi, kk * 32);
    }
    if (hw == 0) {
#pragma unroll
      for (int kk = 0; kk < 2; ++kk) Bo[kk] = frag_gf32(aw2, cl, hi, kk * 32);
    } else {
#pragma unroll
      for (int kk = 0; kk < 2; ++kk) {
#pragma unroll
        for (int v = 0; v < 8; ++v) {
          int k = kk * 32 + kpat(v, hi);
          Bo[kk][2 * v]     = (cl == 0) ? (__bf16)cw2[k]     : (__bf16)0.0f;
          Bo[kk][2 * v + 1] = (cl == 0) ? (__bf16)cw2[k + 1] : (__bf16)0.0f;
        }
      }
    }
    const float bias_feat = base_b[gcol];
    const float bias_a1   = ab1[gcol];
    const float bias_c1   = cb1[gcol];
    const float bias_l    = ab2[cl];
    const float bias_v    = cb2[0];

    issue_x_quarter(x + (size_t)bbase * OBS_D, xsf[0], hw, lane);  // x(0)

    for (int t = 0; t <= T_DIM; ++t) {
      if (t < T_DIM) {
        // ---- x pipeline: my quarter of x(t) landed; kick x(t+1) ----
        wait_async0();
        publish(&c_x, lane);
        if (t + 1 < T_DIM)
          issue_x_quarter(x + ((size_t)(t + 1) * B_DIM + bbase) * OBS_D,
                          xsf[(t + 1) & 1], hw, lane);
        wait_ge(&c_x, 4 * (t + 1));           // whole x(t) tile visible

        // ---- feat(t), up to 4 steps ahead of the recurrence ----
        if (t >= 4) wait_ge(&c_h, 4 * (t - 3)); // feat-ring slot t&3 free
        v8f acc;
#pragma unroll
        for (int i = 0; i < 8; ++i) acc[i] = bias_feat;
#pragma unroll
        for (int kk = 0; kk < 2; ++kk)
          acc = wmma_bf16(frag_lds_f32(xsf[t & 1], cl, hi, kk * 32, 64),
                          Bbase[kk], acc);
        __bf16* fdst = featR[t & 3];
#pragma unroll
        for (int i = 0; i < 8; ++i) {
          int row = i + hi * 8;
          fdst[row * 64 + gcol] = (__bf16)fast_tanh(acc[i]);
        }
        publish(&c_feat, lane);
      }

      if (t >= 1) {
        const int s = t - 1;                  // step whose h we consume
        wait_ge(&c_h, 4 * t);                 // h(s) ready
        if (s >= 2) wait_ge(&c_out, 2 * (s - 1)); // asc slot s&1 retired
        const __bf16* hsl = hR[s & 3];
        v16bf hA[2];
#pragma unroll
        for (int kk = 0; kk < 2; ++kk)
          hA[kk] = frag_lds(hsl, cl, hi, kk * 32, 64);
        v8f aa, cc;
#pragma unroll
        for (int i = 0; i < 8; ++i) { aa[i] = bias_a1; cc[i] = bias_c1; }
#pragma unroll
        for (int kk = 0; kk < 2; ++kk) {
          aa = wmma_bf16(hA[kk], Ba1[kk], aa);
          cc = wmma_bf16(hA[kk], Bc1[kk], cc);
        }
        __bf16* adst = as1R[s & 1];
        __bf16* cdst = cs1R[s & 1];
#pragma unroll
        for (int i = 0; i < 8; ++i) {
          int row = i + hi * 8;
          adst[row * 64 + gcol] = (__bf16)fast_tanh(aa[i]);
          cdst[row * 64 + gcol] = (__bf16)fast_tanh(cc[i]);
        }
        publish(&c_hd, lane);                 // also frees GRU h-ring slot

        if (hw < 2) {                         // output GEMMs + stores
          wait_ge(&c_hd, 4 * t);              // full a1/c1 tile present
          v8f o;
          if (hw == 0) {
#pragma unroll
            for (int i = 0; i < 8; ++i) o[i] = bias_l;
#pragma unroll
            for (int kk = 0; kk < 2; ++kk)
              o = wmma_bf16(frag_lds(as1R[s & 1], cl, hi, kk * 32, 64),
                            Bo[kk], o);
            float* outp = out_logits + (size_t)s * B_DIM * A_D
                                     + (size_t)bbase * A_D;
#pragma unroll
            for (int i = 0; i < 8; ++i)
              outp[(i + hi * 8) * A_D + cl] = o[i];
          } else {
#pragma unroll
            for (int i = 0; i < 8; ++i) o[i] = bias_v;
#pragma unroll
            for (int kk = 0; kk < 2; ++kk)
              o = wmma_bf16(frag_lds(cs1R[s & 1], cl, hi, kk * 32, 64),
                            Bo[kk], o);
            if (cl == 0) {
              float* outp = out_values + (size_t)s * B_DIM + bbase;
#pragma unroll
              for (int i = 0; i < 8; ++i) outp[i + hi * 8] = o[i];
            }
          }
          publish(&c_out, lane);              // asc ring slot retired
        }
      }
    }
  }
}

extern "C" void kernel_launch(void* const* d_in, const int* in_sizes, int n_in,
                              void* d_out, int out_size, void* d_ws, size_t ws_size,
                              hipStream_t stream) {
  const float* x      = (const float*)d_in[0];
  const float* hx     = (const float*)d_in[1];
  const unsigned char* dones = (const unsigned char*)d_in[2];  // jnp bool -> 1B
  const float* base_w = (const float*)d_in[3];
  const float* base_b = (const float*)d_in[4];
  const float* w_ih   = (const float*)d_in[5];
  const float* w_hh   = (const float*)d_in[6];
  const float* b_ih   = (const float*)d_in[7];
  const float* b_hh   = (const float*)d_in[8];
  const float* aw1    = (const float*)d_in[9];
  const float* ab1    = (const float*)d_in[10];
  const float* aw2    = (const float*)d_in[11];
  const float* ab2    = (const float*)d_in[12];
  const float* cw1    = (const float*)d_in[13];
  const float* cb1    = (const float*)d_in[14];
  const float* cw2    = (const float*)d_in[15];
  const float* cb2    = (const float*)d_in[16];

  float* out        = (float*)d_out;
  float* out_logits = out;                                 // T*B*A
  float* out_values = out + (size_t)T_DIM * B_DIM * A_D;   // T*B
  float* out_h      = out_values + (size_t)T_DIM * B_DIM;  // B*GH

  dim3 grid(B_DIM / 16);
  dim3 block(256);
  rac_fused_kernel<<<grid, block, 0, stream>>>(
      x, hx, dones, base_w, base_b, w_ih, w_hh, b_ih, b_hh,
      aw1, ab1, aw2, ab2, cw1, cb1, cw2, cb2,
      out_logits, out_values, out_h);
}
